// Block_65180423685450
// MI455X (gfx1250) — matrix operand order repaired and checked by reading, not compiled, not run on hardware
//
#include <hip/hip_runtime.h>
#include <hip/hip_bf16.h>
#include <math.h>

// ---------------------------------------------------------------------------
// HorNet gnconv block for MI455X (gfx1250), bf16 WMMA GEMMs + fused epilogues.
// Tokens = B*H*W = 16*32*32 = 16384, C = 512.
// All GEMM N dims are multiples of 64 and K dims multiples of 32, so the
// WMMA loops are guard-free (EXEC all-1s as required by the ISA).
// ---------------------------------------------------------------------------

#define TOKENS 16384
#define CDIM   512
#define DWCH   992   // 32+64+128+256+512

typedef __attribute__((ext_vector_type(16))) __bf16 v16bf;
typedef __attribute__((ext_vector_type(8)))  float  v8f;

static __device__ __forceinline__ __bf16 f2bf(float f) {
  unsigned u = __builtin_bit_cast(unsigned, f);
  unsigned r = u + 0x7FFFu + ((u >> 16) & 1u);   // round-to-nearest-even
  unsigned short h = (unsigned short)(r >> 16);
  return __builtin_bit_cast(__bf16, h);
}
static __device__ __forceinline__ float bf2f(__bf16 b) {
  unsigned short h = __builtin_bit_cast(unsigned short, b);
  return __builtin_bit_cast(float, ((unsigned)h) << 16);
}

// --------------------------- fp32 -> bf16 convert ---------------------------
__global__ __launch_bounds__(256) void cvt_f32_bf16(const float* __restrict__ s,
                                                    __bf16* __restrict__ d, int n) {
  int i = blockIdx.x * 256 + threadIdx.x;
  if (i < n) d[i] = f2bf(s[i]);
}

// ------------------------- per-token LayerNorm (C=512) -----------------------
// One wave per token; lane handles channels c = lane + 32*i, i in [0,16).
template<bool NCHW>
__global__ __launch_bounds__(256) void ln_token_kernel(const float* __restrict__ x,
                                                       const float* __restrict__ w,
                                                       const float* __restrict__ b,
                                                       __bf16* __restrict__ out) {
  int gid  = blockIdx.x * 256 + threadIdx.x;
  int m    = gid >> 5;
  int lane = gid & 31;
  if (m >= TOKENS) return;
  int bidx = m >> 10, hw = m & 1023;

  float v[16];
  float s = 0.f;
#pragma unroll
  for (int i = 0; i < 16; ++i) {
    int c = lane + i * 32;
    float xv = NCHW ? x[((size_t)bidx * CDIM + c) * 1024 + hw]
                    : x[(size_t)m * CDIM + c];
    v[i] = xv; s += xv;
  }
#pragma unroll
  for (int o = 16; o >= 1; o >>= 1) s += __shfl_xor(s, o, 32);
  float mean = s * (1.f / 512.f);
  float var = 0.f;
#pragma unroll
  for (int i = 0; i < 16; ++i) { float d0 = v[i] - mean; var += d0 * d0; }
#pragma unroll
  for (int o = 16; o >= 1; o >>= 1) var += __shfl_xor(var, o, 32);
  float inv = rsqrtf(var * (1.f / 512.f) + 1e-6f);
#pragma unroll
  for (int i = 0; i < 16; ++i) {
    int c = lane + i * 32;
    out[(size_t)m * CDIM + c] = f2bf((v[i] - mean) * inv * w[c] + b[c]);
  }
}

// ------------------------- depthwise 7x7 conv (992 ch) ----------------------
__global__ __launch_bounds__(256) void dwconv7x7_kernel(const __bf16* __restrict__ fused,
                                                        const float* __restrict__ dw_w,
                                                        const float* __restrict__ dw_b,
                                                        __bf16* __restrict__ dwb) {
  long long idx = (long long)blockIdx.x * 256 + threadIdx.x;
  if (idx >= (long long)TOKENS * DWCH) return;
  int c = (int)(idx % DWCH);
  int m = (int)(idx / DWCH);
  int bidx = m >> 10, h = (m >> 5) & 31, w = m & 31;
  const float* wk = dw_w + (size_t)c * 49;
  float acc = 0.f;
#pragma unroll
  for (int ky = 0; ky < 7; ++ky) {
    int hh = h + ky - 3;
    if ((unsigned)hh >= 32u) continue;
#pragma unroll
    for (int kx = 0; kx < 7; ++kx) {
      int ww = w + kx - 3;
      if ((unsigned)ww >= 32u) continue;
      int t = (bidx << 10) + (hh << 5) + ww;
      acc += bf2f(fused[(size_t)t * 1024 + 32 + c]) * wk[ky * 7 + kx];
    }
  }
  dwb[idx] = f2bf(acc + dw_b[c]);
}

// --------------------------- y0 = pwa * dw[:, :32] ---------------------------
__global__ __launch_bounds__(256) void mul_gate_kernel(const __bf16* __restrict__ fused,
                                                       const __bf16* __restrict__ dwb,
                                                       __bf16* __restrict__ y0) {
  int i = blockIdx.x * 256 + threadIdx.x;
  if (i >= TOKENS * 32) return;
  int m = i >> 5, c = i & 31;
  y0[i] = f2bf(bf2f(fused[(size_t)m * 1024 + c]) * bf2f(dwb[(size_t)m * DWCH + c]));
}

// ------------------------------ WMMA GEMM -----------------------------------
// out[m][n] = sum_k A[m][k] * W[n][k]  (+ fused epilogue per MODE)
// Each wave computes a 32(M) x 64(N) tile: 2 A-fragments x 4 B-fragments,
// 8 v_wmma_f32_16x16x32_bf16 per 32-wide K step. All loads of a k-step are
// issued before the first WMMA so the compiler can stagger s_wait_loadcnt.
// MODE 0: bf16 out = acc + bias
// MODE 1: bf16 out = (acc + bias) * gate[m][goff + n]              (dw gating)
// MODE 2: f32 x2[m][n] = x_nchw[b][n][hw] + gvec[n]*(acc + bias)   (pout res)
// MODE 3: bf16 out = gelu_exact(acc + bias)                        (fc1)
// MODE 4: f32 out_nchw[b][n][hw] = x2[m][n] + gvec[n]*(acc + bias) (fc2 res)
struct GemmP {
  const __bf16* A;       // [M][lda] bf16 row-major (tokens x K)
  const __bf16* W;       // [N][ldw] bf16 row-major (out_ch x K)
  const float*  bias;    // [N]
  __bf16*       out_bf;  // MODE 0/1/3
  const __bf16* gate;    // MODE 1
  const float*  x_nchw;  // MODE 2
  const float*  gvec;    // MODE 2/4
  float*        x2;      // MODE 2 (write) / MODE 4 (read)
  float*        out_f;   // MODE 4 (NCHW)
  int M, N, K, lda, ldw, ldo, gstride, goff;
};

template<int MODE>
__global__ __launch_bounds__(256) void gemm_wmma(GemmP p) {
  const int lane   = threadIdx.x & 31;
  const int wave   = blockIdx.x * 8 + (threadIdx.x >> 5);
  const int tilesN = p.N >> 6;              // N % 64 == 0 by construction
  const int im     = wave / tilesN;         // 32-row tile
  const int in     = wave % tilesN;         // 64-col tile
  if (im * 32 >= p.M) return;               // wave-uniform

  const int half = lane >> 4;               // K-half selector (A and B)
  const int l15  = lane & 15;

  v8f acc[2][4];
#pragma unroll
  for (int mi = 0; mi < 2; ++mi)
#pragma unroll
    for (int j = 0; j < 4; ++j)
      acc[mi][j] = (v8f){0.f,0.f,0.f,0.f,0.f,0.f,0.f,0.f};

  const __bf16* __restrict__ a0 = p.A + (size_t)(im * 32 + l15)      * p.lda + half * 8;
  const __bf16* __restrict__ a1 = p.A + (size_t)(im * 32 + 16 + l15) * p.lda + half * 8;
  const __bf16* __restrict__ wb = p.W + (size_t)(in * 64 + l15)      * p.ldw + half * 8;
  const size_t wrowstep = (size_t)16 * p.ldw;   // 16 output channels per j-tile

  for (int k0 = 0; k0 < p.K; k0 += 32) {
    union { v16bf v; uint4 q[2]; } af0, af1, bf[4];
    // Issue ALL loads of this k-step first (A pair + 4 B fragments), so
    // waits before each WMMA can be partial instead of loadcnt==0 drains.
    af0.q[0] = *(const uint4*)(a0 + k0);
    af0.q[1] = *(const uint4*)(a0 + k0 + 16);
    af1.q[0] = *(const uint4*)(a1 + k0);
    af1.q[1] = *(const uint4*)(a1 + k0 + 16);
#pragma unroll
    for (int j = 0; j < 4; ++j) {
      const __bf16* wrow = wb + (size_t)j * wrowstep + k0;
      bf[j].q[0] = *(const uint4*)(wrow);
      bf[j].q[1] = *(const uint4*)(wrow + 16);
    }
    if (k0 + 32 < p.K) {
      __builtin_prefetch(a0 + k0 + 32, 0, 3);   // near-cache (WGP) prefetch
      __builtin_prefetch(a1 + k0 + 32, 0, 3);
    }
#pragma unroll
    for (int j = 0; j < 4; ++j) {
      acc[0][j] = __builtin_amdgcn_wmma_f32_16x16x32_bf16(
          false, af0.v, false, bf[j].v, (short)0, acc[0][j], false, false);
      acc[1][j] = __builtin_amdgcn_wmma_f32_16x16x32_bf16(
          false, af1.v, false, bf[j].v, (short)0, acc[1][j], false, false);
    }
  }

#pragma unroll
  for (int mi = 0; mi < 2; ++mi) {
#pragma unroll
    for (int j = 0; j < 4; ++j) {
      int ncol = in * 64 + j * 16 + l15;
      float bv = p.bias[ncol];
#pragma unroll
      for (int r = 0; r < 8; ++r) {
        int m = im * 32 + mi * 16 + half * 8 + r;   // D layout: VGPR r -> row
        float v = acc[mi][j][r] + bv;
        if (MODE == 0) {
          p.out_bf[(size_t)m * p.ldo + ncol] = f2bf(v);
        } else if (MODE == 1) {
          float g = bf2f(p.gate[(size_t)m * p.gstride + p.goff + ncol]);
          p.out_bf[(size_t)m * p.ldo + ncol] = f2bf(v * g);
        } else if (MODE == 2) {
          int bidx = m >> 10, hw = m & 1023;
          float xo = p.x_nchw[((size_t)bidx * CDIM + ncol) * 1024 + hw];
          p.x2[(size_t)m * CDIM + ncol] = xo + p.gvec[ncol] * v;
        } else if (MODE == 3) {
          float gel = 0.5f * v * (1.0f + erff(v * 0.70710678118654752f));
          p.out_bf[(size_t)m * p.ldo + ncol] = f2bf(gel);
        } else { // MODE 4
          int bidx = m >> 10, hw = m & 1023;
          float res = p.x2[(size_t)m * CDIM + ncol] + p.gvec[ncol] * v;
          p.out_f[((size_t)bidx * CDIM + ncol) * 1024 + hw] = res;
        }
      }
    }
  }
}

// ---------------------------------------------------------------------------
extern "C" void kernel_launch(void* const* d_in, const int* in_sizes, int n_in,
                              void* d_out, int out_size, void* d_ws, size_t ws_size,
                              hipStream_t stream) {
  const float* x      = (const float*)d_in[0];
  const float* ln1_w  = (const float*)d_in[1];
  const float* ln1_b  = (const float*)d_in[2];
  const float* pin_w  = (const float*)d_in[3];
  const float* pin_b  = (const float*)d_in[4];
  const float* dw_w   = (const float*)d_in[5];
  const float* dw_b   = (const float*)d_in[6];
  const float* pw_w[4] = {(const float*)d_in[7],  (const float*)d_in[9],
                          (const float*)d_in[11], (const float*)d_in[13]};
  const float* pw_b[4] = {(const float*)d_in[8],  (const float*)d_in[10],
                          (const float*)d_in[12], (const float*)d_in[14]};
  const float* pout_w = (const float*)d_in[15];
  const float* pout_b = (const float*)d_in[16];
  const float* ln2_w  = (const float*)d_in[17];
  const float* ln2_b  = (const float*)d_in[18];
  const float* fc1_w  = (const float*)d_in[19];
  const float* fc1_b  = (const float*)d_in[20];
  const float* fc2_w  = (const float*)d_in[21];
  const float* fc2_b  = (const float*)d_in[22];
  const float* g1     = (const float*)d_in[23];
  const float* g2     = (const float*)d_in[24];
  float* out = (float*)d_out;

  char* ws = (char*)d_ws;
  size_t cur = 0;
  auto alloc = [&](size_t bytes) { size_t o = cur; cur += (bytes + 255) & ~(size_t)255; return o; };

  const int DIMS[5] = {32, 64, 128, 256, 512};

  // bf16 weights
  size_t wb_pin  = alloc((size_t)1024 * 512 * 2);
  size_t wb_pw[4];
  for (int i = 0; i < 4; ++i) wb_pw[i] = alloc((size_t)DIMS[i + 1] * DIMS[i] * 2);
  size_t wb_pout = alloc((size_t)512 * 512 * 2);
  size_t wb_fc1  = alloc((size_t)2048 * 512 * 2);
  size_t wb_fc2  = alloc((size_t)512 * 2048 * 2);

  // activations
  size_t xn_o    = alloc((size_t)TOKENS * CDIM * 2);        // LN out (reused for LN2)
  size_t fused_o = alloc((size_t)TOKENS * 1024 * 2);        // pin output
  size_t dwb_o   = alloc((size_t)TOKENS * DWCH * 2);        // dw conv output
  size_t y_o[5];
  for (int i = 0; i < 5; ++i) y_o[i] = alloc((size_t)TOKENS * DIMS[i] * 2);
  size_t x2_o    = alloc((size_t)TOKENS * CDIM * 4);        // residual-1 (fp32)
  size_t h_o     = fused_o;  // 67.1MB MLP hidden reuses fused+dwb+y0 (dead by then)

  __bf16* WBpin  = (__bf16*)(ws + wb_pin);
  __bf16* WBpout = (__bf16*)(ws + wb_pout);
  __bf16* WBfc1  = (__bf16*)(ws + wb_fc1);
  __bf16* WBfc2  = (__bf16*)(ws + wb_fc2);
  __bf16* XN     = (__bf16*)(ws + xn_o);
  __bf16* FUSED  = (__bf16*)(ws + fused_o);
  __bf16* DWB    = (__bf16*)(ws + dwb_o);
  __bf16* H      = (__bf16*)(ws + h_o);
  float*  X2     = (float*)(ws + x2_o);

  auto cvt = [&](const float* s, size_t off, int n) {
    cvt_f32_bf16<<<(n + 255) / 256, 256, 0, stream>>>(s, (__bf16*)(ws + off), n);
  };
  cvt(pin_w, wb_pin, 1024 * 512);
  for (int i = 0; i < 4; ++i) cvt(pw_w[i], wb_pw[i], DIMS[i + 1] * DIMS[i]);
  cvt(pout_w, wb_pout, 512 * 512);
  cvt(fc1_w, wb_fc1, 2048 * 512);
  cvt(fc2_w, wb_fc2, 512 * 2048);

  auto launch_gemm = [&](int mode, GemmP p) {
    int tilesN = p.N >> 6;
    int waves  = (p.M / 32) * tilesN;
    int blocks = (waves + 7) / 8;
    switch (mode) {
      case 0: gemm_wmma<0><<<blocks, 256, 0, stream>>>(p); break;
      case 1: gemm_wmma<1><<<blocks, 256, 0, stream>>>(p); break;
      case 2: gemm_wmma<2><<<blocks, 256, 0, stream>>>(p); break;
      case 3: gemm_wmma<3><<<blocks, 256, 0, stream>>>(p); break;
      default: gemm_wmma<4><<<blocks, 256, 0, stream>>>(p); break;
    }
  };

  // 1) LN1 (channels-first) -> bf16 tokens
  ln_token_kernel<true><<<TOKENS * 32 / 256, 256, 0, stream>>>(x, ln1_w, ln1_b, XN);

  // 2) pin: 512 -> 1024
  { GemmP p{}; p.A = XN; p.W = WBpin; p.bias = pin_b; p.out_bf = FUSED;
    p.M = TOKENS; p.N = 1024; p.K = 512; p.lda = 512; p.ldw = 512; p.ldo = 1024;
    launch_gemm(0, p); }

  // 3) depthwise 7x7 over 992 channels
  { long long tot = (long long)TOKENS * DWCH;
    dwconv7x7_kernel<<<(unsigned)((tot + 255) / 256), 256, 0, stream>>>(FUSED, dw_w, dw_b, DWB); }

  // 4) y0 = pwa * dw[:, 0:32]
  mul_gate_kernel<<<TOKENS * 32 / 256, 256, 0, stream>>>(FUSED, DWB, (__bf16*)(ws + y_o[0]));

  // 5) recursive gated pointwise convs; dw slice offsets = cumsum(DIMS)
  const int goffs[4] = {32, 96, 224, 480};
  for (int i = 0; i < 4; ++i) {
    GemmP p{}; p.A = (__bf16*)(ws + y_o[i]); p.W = (__bf16*)(ws + wb_pw[i]);
    p.bias = pw_b[i]; p.out_bf = (__bf16*)(ws + y_o[i + 1]);
    p.gate = DWB; p.gstride = DWCH; p.goff = goffs[i];
    p.M = TOKENS; p.N = DIMS[i + 1]; p.K = DIMS[i];
    p.lda = DIMS[i]; p.ldw = DIMS[i]; p.ldo = DIMS[i + 1];
    launch_gemm(1, p);
  }

  // 6) pout + residual-1: x2 = x + g1 * (pout(y4) + b)
  { GemmP p{}; p.A = (__bf16*)(ws + y_o[4]); p.W = WBpout; p.bias = pout_b;
    p.x_nchw = x; p.gvec = g1; p.x2 = X2;
    p.M = TOKENS; p.N = 512; p.K = 512; p.lda = 512; p.ldw = 512;
    launch_gemm(2, p); }

  // 7) LN2 (token-major fp32 in) -> bf16 tokens (reuse XN)
  ln_token_kernel<false><<<TOKENS * 32 / 256, 256, 0, stream>>>(X2, ln2_w, ln2_b, XN);

  // 8) fc1 + exact GELU: 512 -> 2048
  { GemmP p{}; p.A = XN; p.W = WBfc1; p.bias = fc1_b; p.out_bf = H;
    p.M = TOKENS; p.N = 2048; p.K = 512; p.lda = 512; p.ldw = 512; p.ldo = 2048;
    launch_gemm(3, p); }

  // 9) fc2 + residual-2 -> NCHW fp32 output
  { GemmP p{}; p.A = H; p.W = WBfc2; p.bias = fc2_b;
    p.x2 = X2; p.gvec = g2; p.out_f = out;
    p.M = TOKENS; p.N = 512; p.K = 2048; p.lda = 2048; p.ldw = 2048;
    launch_gemm(4, p); }
}
